// GPT_62732292325712
// MI455X (gfx1250) — matrix operand order repaired and checked by reading, not compiled, not run on hardware
//
#include <hip/hip_runtime.h>
#include <hip/hip_bf16.h>
#include <stdint.h>

// ---------------- model constants ----------------
#define SEQ   1024
#define EMB   512
#define NHEAD 8
#define HDIM  64
#define FFD   2048
#define VOCAB 32000
#define NLAYER 4
#define NTOK  2048      // B * SEQ

// ---------------- WMMA types ----------------
typedef __bf16 bf16_t;
typedef bf16_t v16bf __attribute__((ext_vector_type(16)));
typedef float  v8f   __attribute__((ext_vector_type(8)));

union Frag16 {
  v16bf v;
  unsigned short s[16];
  uint4 q[2];
};

__device__ __forceinline__ unsigned short f2bf(float f) {
  union { float f; unsigned int u; } c; c.f = f;
  unsigned int u = c.u;
  u += 0x7FFFu + ((u >> 16) & 1u);        // round-to-nearest-even
  return (unsigned short)(u >> 16);
}
__device__ __forceinline__ unsigned int pack2(float lo, float hi) {
  return (unsigned int)f2bf(lo) | ((unsigned int)f2bf(hi) << 16);
}
__device__ __forceinline__ v8f vzero8() {
  v8f v;
#pragma unroll
  for (int i = 0; i < 8; ++i) v[i] = 0.0f;
  return v;
}
// pack 8 consecutive floats -> uint4 of bf16 pairs
__device__ __forceinline__ uint4 pack8(const float* p) {
  float4 a = *(const float4*)p;
  float4 b = *(const float4*)(p + 4);
  uint4 w;
  w.x = pack2(a.x, a.y); w.y = pack2(a.z, a.w);
  w.z = pack2(b.x, b.y); w.w = pack2(b.z, b.w);
  return w;
}
// fill 8 consecutive fragment elements from 8 consecutive floats (scaled)
__device__ __forceinline__ void fill8(Frag16& f, int e0, const float* p, float scale) {
  float4 a = *(const float4*)p;
  float4 b = *(const float4*)(p + 4);
  f.s[e0+0] = f2bf(a.x*scale); f.s[e0+1] = f2bf(a.y*scale);
  f.s[e0+2] = f2bf(a.z*scale); f.s[e0+3] = f2bf(a.w*scale);
  f.s[e0+4] = f2bf(b.x*scale); f.s[e0+5] = f2bf(b.y*scale);
  f.s[e0+6] = f2bf(b.z*scale); f.s[e0+7] = f2bf(b.w*scale);
}

// ================= embedding: h = tok_emb[x] + pos_emb =================
__global__ __launch_bounds__(256) void embed_kernel(
    const int* __restrict__ x, const float* __restrict__ tok,
    const float* __restrict__ pos, float* __restrict__ h) {
  int idx = blockIdx.x * 256 + threadIdx.x;      // over NTOK*EMB
  int m = idx >> 9, e = idx & 511;
  int t = m & (SEQ - 1);
  h[idx] = tok[(size_t)x[m] * EMB + e] + pos[(size_t)t * EMB + e];
}

// ========== repack per-head weight [H,E,D] -> [E, H*D] row-major ==========
__global__ __launch_bounds__(256) void repack_kernel(
    const float* __restrict__ W, float* __restrict__ Wr) {
  int idx = blockIdx.x * 256 + threadIdx.x;      // over EMB*EMB
  int e = idx >> 9, n = idx & 511;
  Wr[idx] = W[(size_t)(n >> 6) * (EMB * HDIM) + (size_t)e * HDIM + (n & 63)];
}

// ================= generic bf16-WMMA GEMM =================
// C[M,N] = act( A[M,K] @ B[K,N] + bias[N] ), f32 in/out, bf16 WMMA, f32 acc.
// Block tile 64x64, K-step 64, 8 waves: wave(wr,wc) owns a 16x32 strip
// (two 16x16 subtiles) -> 4 WMMAs between each barrier pair.
__global__ __launch_bounds__(256) void gemm_bias_kernel(
    const float* __restrict__ A, const float* __restrict__ B,
    const float* __restrict__ bias, float* __restrict__ C,
    int M, int N, int K, int relu) {
  __shared__ __align__(16) unsigned short As[64 * 64];   // [row][k] bf16
  __shared__ __align__(16) unsigned short Bt[64 * 64];   // [n][k]  bf16 (transposed)
  const int tid  = threadIdx.x;
  const int lane = tid & 31;
  const int wave = tid >> 5;
  const int hi   = (lane >> 4) & 1;
  const int l16  = lane & 15;
  const int m0 = blockIdx.y * 64;
  const int n0 = blockIdx.x * 64;
  const int wr = wave & 3;          // row subtile 0..3
  const int wc = wave >> 2;         // col strip  0..1 (covers 32 cols)

  v8f acc[2];
  acc[0] = vzero8(); acc[1] = vzero8();

  const int srow = tid >> 2;               // 0..63
  const int scol = (tid & 3) * 16;         // 0,16,32,48

  for (int k0 = 0; k0 < K; k0 += 64) {
    // ---- stage A tile (64x64): 16 contiguous floats per thread ----
    {
      const float* ga = A + (size_t)(m0 + srow) * K + k0 + scol;
      if (k0 + 64 < K) __builtin_prefetch(ga + 64, 0, 0);
      *(uint4*)&As[srow * 64 + scol]     = pack8(ga);
      *(uint4*)&As[srow * 64 + scol + 8] = pack8(ga + 8);
    }
    // ---- stage B tile (64x64) transposed: 16 floats per thread ----
    {
      const float* gb = B + (size_t)(k0 + srow) * N + n0 + scol;
#pragma unroll
      for (int j = 0; j < 16; j += 4) {
        float4 p = *(const float4*)(gb + j);
        Bt[(scol + j + 0) * 64 + srow] = f2bf(p.x);
        Bt[(scol + j + 1) * 64 + srow] = f2bf(p.y);
        Bt[(scol + j + 2) * 64 + srow] = f2bf(p.z);
        Bt[(scol + j + 3) * 64 + srow] = f2bf(p.w);
      }
    }
    __syncthreads();
    // ---- 2 k-substeps x 2 n-subtiles = 4 WMMAs per barrier round ----
#pragma unroll
    for (int ks = 0; ks < 64; ks += 32) {
      Frag16 a;
      {
        const unsigned short* pa = &As[(wr * 16 + l16) * 64 + ks + hi * 8];
        a.q[0] = *(const uint4*)pa;          // elems 0..7 : k = ks + 8*hi + i
        a.q[1] = *(const uint4*)(pa + 16);   // elems 8..15: k = ks + 8*hi + 16 + (i-8)
      }
#pragma unroll
      for (int t = 0; t < 2; ++t) {
        Frag16 b;
        const unsigned short* pb = &Bt[(wc * 32 + t * 16 + l16) * 64 + ks + hi * 8];
        b.q[0] = *(const uint4*)pb;          // elems 0..7 : k = ks + 8*hi + i
        b.q[1] = *(const uint4*)(pb + 16);   // elems 8..15: k = ks + 8*hi + 16 + (i-8)
        acc[t] = __builtin_amdgcn_wmma_f32_16x16x32_bf16(false, a.v, false, b.v,
                                                         (short)0, acc[t], false, false);
      }
    }
    __syncthreads();
  }
  // ---- epilogue: bias + optional relu; C layout: row = r + 8*hi, col = lane%16 ----
#pragma unroll
  for (int t = 0; t < 2; ++t) {
    const int col = n0 + wc * 32 + t * 16 + l16;
    const float bv = bias[col];
#pragma unroll
    for (int r = 0; r < 8; ++r) {
      int row = m0 + wr * 16 + r + 8 * hi;
      float v = acc[t][r] + bv;
      if (relu) v = fmaxf(v, 0.0f);
      C[(size_t)row * N + col] = v;
    }
  }
}

// ================= flash attention, one wave per (b, h, q-tile16) =================
// Q,K,V,O: [NTOK, EMB] with column = h*64 + d.
__global__ __launch_bounds__(128) void attention_kernel(
    const float* __restrict__ Q, const float* __restrict__ K,
    const float* __restrict__ V, float* __restrict__ O) {
  __shared__ __align__(16) unsigned short PtS[4 * 16 * 32];  // P tile per wave
  __shared__ __align__(16) unsigned short VtS[4 * 64 * 32];  // V^T tile per wave
  const int tid   = threadIdx.x;
  const int lane  = tid & 31;
  const int wslot = tid >> 5;
  const int hi    = (lane >> 4) & 1;
  const int l16   = lane & 15;
  unsigned short* pt = PtS + wslot * (16 * 32);
  unsigned short* vt = VtS + wslot * (64 * 32);

  int task = blockIdx.x * 4 + wslot;     // 0..1023
  int qt = task & 63;
  int h  = (task >> 6) & 7;
  int b  = task >> 9;
  const size_t base = ((size_t)b * SEQ) * EMB + (size_t)h * HDIM;

  // ---- Q fragments (2 d-chunks of 32), 1/sqrt(D)=0.125 folded in ----
  Frag16 qf[2];
  {
    const float* qp = Q + base + (size_t)(qt * 16 + l16) * EMB;
#pragma unroll
    for (int c = 0; c < 2; ++c) {
      int d0 = c * 32 + hi * 8;
      fill8(qf[c], 0, qp + d0,      0.125f);   // k = 8*hi + i
      fill8(qf[c], 8, qp + d0 + 16, 0.125f);   // k = 8*hi + 16 + (i-8)
    }
  }

  v8f oacc[4]; float rm[8], rs[8];
#pragma unroll
  for (int g = 0; g < 4; ++g) oacc[g] = vzero8();
#pragma unroll
  for (int r = 0; r < 8; ++r) { rm[r] = -1e30f; rs[r] = 0.0f; }

  const int nchunks = (qt * 16 + 15) / 32 + 1;     // causal: keys <= q
  for (int kc = 0; kc < nchunks; ++kc) {
    const int kb = kc * 32;
    // ---- scores: two 16-key tiles, contraction over D=64 (2 WMMAs each) ----
    v8f s0 = vzero8(), s1 = vzero8();
#pragma unroll
    for (int c = 0; c < 2; ++c) {
      Frag16 bk;
      const float* kp0 = K + base + (size_t)(kb + l16) * EMB + c * 32 + hi * 8;
      fill8(bk, 0, kp0, 1.0f); fill8(bk, 8, kp0 + 16, 1.0f);
      s0 = __builtin_amdgcn_wmma_f32_16x16x32_bf16(false, qf[c].v, false, bk.v,
                                                   (short)0, s0, false, false);
      const float* kp1 = K + base + (size_t)(kb + 16 + l16) * EMB + c * 32 + hi * 8;
      fill8(bk, 0, kp1, 1.0f); fill8(bk, 8, kp1 + 16, 1.0f);
      s1 = __builtin_amdgcn_wmma_f32_16x16x32_bf16(false, qf[c].v, false, bk.v,
                                                   (short)0, s1, false, false);
    }
    // ---- causal mask + online softmax (row = r + 8*hi, col = lane%16) ----
    const int key0 = kb + l16, key1 = kb + 16 + l16;
#pragma unroll
    for (int r = 0; r < 8; ++r) {
      const int qrow = qt * 16 + r + 8 * hi;
      float a = (key0 <= qrow) ? s0[r] : -1e30f;
      float bb = (key1 <= qrow) ? s1[r] : -1e30f;
      float mx = fmaxf(a, bb);
#pragma unroll
      for (int msk = 8; msk >= 1; msk >>= 1)
        mx = fmaxf(mx, __shfl_xor(mx, msk, 32));   // reduce within 16-lane half
      const float nm = fmaxf(rm[r], mx);
      const float fac = __expf(rm[r] - nm);
      const float p0 = __expf(a - nm);
      const float p1 = __expf(bb - nm);
      s0[r] = p0; s1[r] = p1;
      float sum = p0 + p1;
#pragma unroll
      for (int msk = 8; msk >= 1; msk >>= 1)
        sum += __shfl_xor(sum, msk, 32);
      rs[r] = rs[r] * fac + sum;
      rm[r] = nm;
#pragma unroll
      for (int g = 0; g < 4; ++g) oacc[g][r] *= fac;
    }
    // ---- stage P (C-layout -> LDS [m][kk]) and V^T ([d][key]) ----
#pragma unroll
    for (int r = 0; r < 8; ++r) {
      const int m = r + 8 * hi;
      pt[m * 32 + l16]      = f2bf(s0[r]);
      pt[m * 32 + 16 + l16] = f2bf(s1[r]);
    }
    {
      const float* vp = V + base + (size_t)(kb + lane) * EMB;
#pragma unroll
      for (int d = 0; d < 64; d += 4) {
        float4 p = *(const float4*)(vp + d);
        vt[(d + 0) * 32 + lane] = f2bf(p.x);
        vt[(d + 1) * 32 + lane] = f2bf(p.y);
        vt[(d + 2) * 32 + lane] = f2bf(p.z);
        vt[(d + 3) * 32 + lane] = f2bf(p.w);
      }
    }
    asm volatile("" ::: "memory");  // wave-synchronous LDS: DS ops in-order per ISA
    // ---- P@V: A = P[16x32 keys], 4 WMMAs over d-groups of 16 ----
    Frag16 pa, vb;
    {
      const unsigned short* pp = &pt[l16 * 32 + hi * 8];
      pa.q[0] = *(const uint4*)pp;
      pa.q[1] = *(const uint4*)(pp + 16);
    }
#pragma unroll
    for (int g = 0; g < 4; ++g) {
      const unsigned short* vp2 = &vt[(g * 16 + l16) * 32 + hi * 8];
      vb.q[0] = *(const uint4*)vp2;
      vb.q[1] = *(const uint4*)(vp2 + 16);
      oacc[g] = __builtin_amdgcn_wmma_f32_16x16x32_bf16(false, pa.v, false, vb.v,
                                                        (short)0, oacc[g], false, false);
    }
    asm volatile("" ::: "memory");
  }
  // ---- normalize & store (concat-heads layout) ----
#pragma unroll
  for (int g = 0; g < 4; ++g) {
    const int d = g * 16 + l16;
#pragma unroll
    for (int r = 0; r < 8; ++r) {
      const int qrow = qt * 16 + r + 8 * hi;
      O[base + (size_t)qrow * EMB + d] = oacc[g][r] / rs[r];
    }
  }
}

// ================= residual add + layernorm, one wave per row =================
__global__ __launch_bounds__(256) void add_ln_kernel(
    const float* __restrict__ A, const float* __restrict__ B,
    const float* __restrict__ g, const float* __restrict__ bt,
    float* __restrict__ O) {
  const int lane = threadIdx.x & 31;
  const int row  = blockIdx.x * 8 + (threadIdx.x >> 5);
  const float* a = A + (size_t)row * EMB;
  const float* b = B + (size_t)row * EMB;
  float x[16]; float s = 0.0f;
#pragma unroll
  for (int j = 0; j < 16; ++j) { x[j] = a[lane + j * 32] + b[lane + j * 32]; s += x[j]; }
#pragma unroll
  for (int m = 16; m >= 1; m >>= 1) s += __shfl_xor(s, m, 32);
  const float mean = s * (1.0f / EMB);
  float vs = 0.0f;
#pragma unroll
  for (int j = 0; j < 16; ++j) { float d = x[j] - mean; vs += d * d; }
#pragma unroll
  for (int m = 16; m >= 1; m >>= 1) vs += __shfl_xor(vs, m, 32);
  const float rstd = rsqrtf(vs * (1.0f / EMB) + 1e-5f);
#pragma unroll
  for (int j = 0; j < 16; ++j) {
    const int e = lane + j * 32;
    O[(size_t)row * EMB + e] = (x[j] - mean) * rstd * g[e] + bt[e];
  }
}

// ================= host orchestration =================
extern "C" void kernel_launch(void* const* d_in, const int* in_sizes, int n_in,
                              void* d_out, int out_size, void* d_ws, size_t ws_size,
                              hipStream_t stream) {
  (void)in_sizes; (void)n_in; (void)out_size; (void)ws_size;
  const int*   x    = (const int*)  d_in[0];
  const float* tok  = (const float*)d_in[1];
  const float* pos  = (const float*)d_in[2];
  const float* Wq   = (const float*)d_in[3];
  const float* bq   = (const float*)d_in[4];
  const float* Wk   = (const float*)d_in[5];
  const float* bk   = (const float*)d_in[6];
  const float* Wv   = (const float*)d_in[7];
  const float* bv   = (const float*)d_in[8];
  const float* Wo   = (const float*)d_in[9];
  const float* bo   = (const float*)d_in[10];
  const float* W1   = (const float*)d_in[11];
  const float* b1   = (const float*)d_in[12];
  const float* W2   = (const float*)d_in[13];
  const float* b2   = (const float*)d_in[14];
  const float* g1   = (const float*)d_in[15];
  const float* be1  = (const float*)d_in[16];
  const float* g2   = (const float*)d_in[17];
  const float* be2  = (const float*)d_in[18];
  const float* Wout = (const float*)d_in[19];
  const float* bout = (const float*)d_in[20];
  float* out = (float*)d_out;

  // workspace layout (floats); total ~12.3M floats (~49 MB)
  const size_t TE = (size_t)NTOK * EMB;          // 1,048,576
  float* ws   = (float*)d_ws;
  float* h    = ws;
  float* ln1  = h    + TE;
  float* tmp  = ln1  + TE;
  float* Qb   = tmp  + TE;
  float* Kb   = Qb   + TE;
  float* Vb   = Kb   + TE;
  float* conc = Vb   + TE;
  float* ff1  = conc + TE;                       // NTOK*FFD = 4M floats
  float* wrq  = ff1  + (size_t)NTOK * FFD;
  float* wrk  = wrq  + (size_t)EMB * EMB;
  float* wrv  = wrk  + (size_t)EMB * EMB;

  embed_kernel<<<(NTOK * EMB) / 256, 256, 0, stream>>>(x, tok, pos, h);

  const dim3 gE(EMB / 64, NTOK / 64);    // GEMM -> [NTOK, EMB]
  const dim3 gF(FFD / 64, NTOK / 64);    // GEMM -> [NTOK, FFD]
  const dim3 gV(VOCAB / 64, NTOK / 64);  // GEMM -> [NTOK, VOCAB]

  for (int l = 0; l < NLAYER; ++l) {
    const size_t wOff = (size_t)l * EMB * EMB;     // per-layer [H,E,D] == E*E elems
    repack_kernel<<<(EMB * EMB) / 256, 256, 0, stream>>>(Wq + wOff, wrq);
    repack_kernel<<<(EMB * EMB) / 256, 256, 0, stream>>>(Wk + wOff, wrk);
    repack_kernel<<<(EMB * EMB) / 256, 256, 0, stream>>>(Wv + wOff, wrv);

    gemm_bias_kernel<<<gE, 256, 0, stream>>>(h, wrq, bq + l * EMB, Qb, NTOK, EMB, EMB, 0);
    gemm_bias_kernel<<<gE, 256, 0, stream>>>(h, wrk, bk + l * EMB, Kb, NTOK, EMB, EMB, 0);
    gemm_bias_kernel<<<gE, 256, 0, stream>>>(h, wrv, bv + l * EMB, Vb, NTOK, EMB, EMB, 0);

    attention_kernel<<<256, 128, 0, stream>>>(Qb, Kb, Vb, conc);

    gemm_bias_kernel<<<gE, 256, 0, stream>>>(conc, Wo + wOff, bo + l * EMB, tmp,
                                             NTOK, EMB, EMB, 0);
    add_ln_kernel<<<NTOK / 8, 256, 0, stream>>>(h, tmp, g1 + l * EMB, be1 + l * EMB, ln1);

    gemm_bias_kernel<<<gF, 256, 0, stream>>>(ln1, W1 + (size_t)l * EMB * FFD,
                                             b1 + l * FFD, ff1, NTOK, FFD, EMB, 1);
    gemm_bias_kernel<<<gE, 256, 0, stream>>>(ff1, W2 + (size_t)l * FFD * EMB,
                                             b2 + l * EMB, tmp, NTOK, EMB, FFD, 0);
    add_ln_kernel<<<NTOK / 8, 256, 0, stream>>>(ln1, tmp, g2 + l * EMB, be2 + l * EMB, h);
  }

  gemm_bias_kernel<<<gV, 256, 0, stream>>>(h, Wout, bout, out, NTOK, VOCAB, EMB, 0);
}
